// CausalSelfAttention_79285096284305
// MI455X (gfx1250) — matrix operand order repaired and checked by reading, not compiled, not run on hardware
//
#include <hip/hip_runtime.h>
#include <hip/hip_bf16.h>

// CausalSelfAttention for MI455X (gfx1250), wave32 + v_wmma_f32_16x16x32_f16.
//
// Shapes: B=128, T=256, T_past=256, D=128, H=4, Dh=32, T_full=512.
// d_out layout (flat, fp32): [proj_out (B,T,128)] [k (B,H,512,32)] [v (B,H,512,32)]
// d_ws: q scratch (B,H,T,32) fp32 = 16 MiB, attn scratch (B,T,128) fp32 = 16 MiB.

#define B_SZ 128
#define T_SZ 256
#define TPAST 256
#define TFULL 512
#define DMODEL 128
#define NHEAD 4
#define DHEAD 32
#define ATT_SCALE 0.17677669529663687f /* 1/sqrt(32) */

typedef __attribute__((ext_vector_type(16))) _Float16 v16h;
typedef __attribute__((ext_vector_type(8)))  float    v8f;

__device__ __forceinline__ v8f wmma_f16(v16h a, v16h b, v8f c) {
  // D = A(16x32 f16) * B(32x16 f16) + C(16x16 f32)
  return __builtin_amdgcn_wmma_f32_16x16x32_f16(
      /*neg_a=*/false, a, /*neg_b=*/false, b,
      /*c_mod=*/(short)0, c, /*reuse_a=*/false, /*reuse_b=*/false);
}

// A fragment: 16x32 (MxK) f16 from row-major fp32 matrix (row stride ld floats).
// lane lp = row m, lh selects contiguous K-half [16*lh, 16*lh+16).
__device__ __forceinline__ v16h load_a_f32(const float* __restrict__ base, int ld,
                                           int lp, int lh, int k0, float scale) {
  v16h a;
  const float* p = base + (size_t)lp * ld + k0 + lh * 16;
#pragma unroll
  for (int h = 0; h < 8; ++h) a[h] = (_Float16)(p[h] * scale);
#pragma unroll
  for (int h = 0; h < 8; ++h) a[8 + h] = (_Float16)(p[8 + h] * scale);
  return a;
}

// B fragment: 32x16 (KxN) f16 from row-major fp32 matrix (row stride ld floats),
// covering rows [k0, k0+32), cols [n0, n0+16). lane lp = col, lh = K-half.
__device__ __forceinline__ v16h load_b_f32(const float* __restrict__ base, int ld,
                                           int n, int lh, int k0) {
  v16h b;
  const float* p = base + (size_t)(k0 + lh * 16) * ld + n;
#pragma unroll
  for (int h = 0; h < 16; ++h) b[h] = (_Float16)(p[(size_t)h * ld]);
  return b;
}

// ---------------------------------------------------------------------------
// Kernel 1: copy past_k / past_v into rows [0,256) of each (b,h) slab of k/v out
// ---------------------------------------------------------------------------
__global__ void copy_past_kernel(const float4* __restrict__ pk,
                                 const float4* __restrict__ pv,
                                 float4* __restrict__ ko,
                                 float4* __restrict__ vo) {
  int tid = blockIdx.x * blockDim.x + threadIdx.x;   // 0 .. 1048575
  const float4* src = blockIdx.y ? pv : pk;
  float4* dst = blockIdx.y ? vo : ko;
  int bh  = tid >> 11;          // 2048 float4 per (b,h) past-slab
  int rem = tid & 2047;
  dst[(size_t)bh * (TFULL * DHEAD / 4) + rem] = src[tid];
}

// ---------------------------------------------------------------------------
// Kernel 2: qkv = x @ W_qkv  (32768x128 @ 128x384), scatter q / k_new / v_new
// grid (2048, 24), block 32 (one wave per 16x16 tile)
// ---------------------------------------------------------------------------
__global__ __launch_bounds__(32) void qkv_gemm_kernel(
    const float* __restrict__ x, const float* __restrict__ W,
    float* __restrict__ q_s, float* __restrict__ k_out, float* __restrict__ v_out) {
  int rt = blockIdx.x;                 // row tile: 0..2047
  int ct = blockIdx.y;                 // col tile: 0..23
  int lane = threadIdx.x & 31;
  int lp = lane & 15, lh = lane >> 4;
  int row0 = rt * 16;
  int col0 = ct * 16;

  v8f c = {};
#pragma unroll
  for (int kc = 0; kc < 4; ++kc) {
    v16h a = load_a_f32(x + (size_t)row0 * DMODEL, DMODEL, lp, lh, kc * 32, 1.0f);
    v16h b = load_b_f32(W, 3 * DMODEL, col0 + lp, lh, kc * 32);
    c = wmma_f16(a, b, c);
  }

  int sect = ct >> 3;                  // 0=q, 1=k_new, 2=v_new (uniform per block)
  int cd   = col0 + lp - sect * DMODEL;
  int head = cd >> 5, d = cd & 31;
#pragma unroll
  for (int i = 0; i < 8; ++i) {
    int r  = row0 + i + lh * 8;        // global row = b*256 + t
    int bb = r >> 8, t = r & 255;
    float val = c[i];
    if (sect == 0) {
      q_s[(((size_t)(bb * NHEAD + head) * T_SZ) + t) * DHEAD + d] = val;
    } else {
      size_t idx = (((size_t)(bb * NHEAD + head) * TFULL) + TPAST + t) * DHEAD + d;
      if (sect == 1) k_out[idx] = val; else v_out[idx] = val;
    }
  }
}

// ---------------------------------------------------------------------------
// Kernel 3: flash attention per (b,h, query-tile). grid (512, 16), block 32.
// S = (Q*scale) @ K^T (one WMMA per 16 keys, K-dim = Dh = 32), online softmax,
// O += P @ V (LDS bounce for C-layout -> A-layout, two WMMAs per 32-key chunk).
// ---------------------------------------------------------------------------
__global__ __launch_bounds__(32) void attention_kernel(
    const float* __restrict__ q_s, const float* __restrict__ k_full,
    const float* __restrict__ v_full, float* __restrict__ attn_out) {
  int bh = blockIdx.x;                 // 0..511 = b*4 + h
  int qt = blockIdx.y;                 // 0..15
  int b  = bh >> 2, h = bh & 3;
  int lane = threadIdx.x & 31;
  int lp = lane & 15, lh = lane >> 4;

  __shared__ _Float16 pLds[16][32];

  // Q A-fragment (scale folded in)
  const float* qrow = q_s + ((size_t)bh * T_SZ + qt * 16 + lp) * DHEAD;
  v16h aq;
#pragma unroll
  for (int hh = 0; hh < 8; ++hh) aq[hh]     = (_Float16)(qrow[lh * 16 + hh] * ATT_SCALE);
#pragma unroll
  for (int hh = 0; hh < 8; ++hh) aq[8 + hh] = (_Float16)(qrow[lh * 16 + 8 + hh] * ATT_SCALE);

  const float* kb = k_full + (size_t)bh * TFULL * DHEAD;
  const float* vb = v_full + (size_t)bh * TFULL * DHEAD;

  float rmax[8], rsum[8];
  v8f o0 = {}, o1 = {};
#pragma unroll
  for (int i = 0; i < 8; ++i) { rmax[i] = -1e30f; rsum[i] = 0.0f; }

  for (int ch = 0; ch < TFULL / 32; ++ch) {
    int key0 = ch * 32;
    // K^T B-fragments: B[kk][n] = K[key0+n][kk]; contiguous 16-float runs per lane.
    v16h bk0, bk1;
    const float* kp0 = kb + (size_t)(key0 + lp) * DHEAD + lh * 16;
    const float* kp1 = kb + (size_t)(key0 + 16 + lp) * DHEAD + lh * 16;
#pragma unroll
    for (int hh = 0; hh < 16; ++hh) { bk0[hh] = (_Float16)kp0[hh]; bk1[hh] = (_Float16)kp1[hh]; }
    v8f zc = {};
    v8f s0 = wmma_f16(aq, bk0, zc);
    v8f s1 = wmma_f16(aq, bk1, zc);

    int kpos0 = key0 + lp;
    int kpos1 = key0 + 16 + lp;
    float p0[8], p1[8];
#pragma unroll
    for (int i = 0; i < 8; ++i) {
      int qpos = TPAST + qt * 16 + i + lh * 8;
      float x0 = (kpos0 <= qpos) ? s0[i] : -1e30f;
      float x1 = (kpos1 <= qpos) ? s1[i] : -1e30f;
      float t = fmaxf(x0, x1);
      t = fmaxf(t, __shfl_xor(t, 1, 32));
      t = fmaxf(t, __shfl_xor(t, 2, 32));
      t = fmaxf(t, __shfl_xor(t, 4, 32));
      t = fmaxf(t, __shfl_xor(t, 8, 32));
      float nm = fmaxf(rmax[i], t);
      float sf = __expf(rmax[i] - nm);
      rmax[i] = nm;
      float e0 = (kpos0 <= qpos) ? __expf(x0 - nm) : 0.0f;
      float e1 = (kpos1 <= qpos) ? __expf(x1 - nm) : 0.0f;
      float ps = e0 + e1;
      ps += __shfl_xor(ps, 1, 32);
      ps += __shfl_xor(ps, 2, 32);
      ps += __shfl_xor(ps, 4, 32);
      ps += __shfl_xor(ps, 8, 32);
      rsum[i] = rsum[i] * sf + ps;
      o0[i] *= sf;
      o1[i] *= sf;
      p0[i] = e0;
      p1[i] = e1;
    }

    // C-layout -> A-layout via LDS (single wave; DS ops are in-order per wave).
#pragma unroll
    for (int i = 0; i < 8; ++i) {
      pLds[i + lh * 8][lp]      = (_Float16)p0[i];
      pLds[i + lh * 8][lp + 16] = (_Float16)p1[i];
    }
    v16h ap;
#pragma unroll
    for (int hh = 0; hh < 8; ++hh) ap[hh]     = pLds[lp][lh * 16 + hh];
#pragma unroll
    for (int hh = 0; hh < 8; ++hh) ap[8 + hh] = pLds[lp][lh * 16 + 8 + hh];

    // V B-fragments (cols 0-15 and 16-31 of Dh)
    v16h bv0, bv1;
    const float* vp = vb + (size_t)(key0 + lh * 16) * DHEAD;
#pragma unroll
    for (int hh = 0; hh < 16; ++hh) {
      bv0[hh] = (_Float16)vp[(size_t)hh * DHEAD + lp];
      bv1[hh] = (_Float16)vp[(size_t)hh * DHEAD + 16 + lp];
    }
    o0 = wmma_f16(ap, bv0, o0);
    o1 = wmma_f16(ap, bv1, o1);
  }

  // Epilogue: normalize, write attn (B, T, 128) with cols [h*32, h*32+32)
#pragma unroll
  for (int i = 0; i < 8; ++i) {
    float inv = 1.0f / rsum[i];
    size_t r = (size_t)b * T_SZ + qt * 16 + i + lh * 8;
    attn_out[r * DMODEL + h * DHEAD + lp]      = o0[i] * inv;
    attn_out[r * DMODEL + h * DHEAD + 16 + lp] = o1[i] * inv;
  }
}

// ---------------------------------------------------------------------------
// Kernel 4: out = attn @ W_proj (32768x128 @ 128x128). grid (2048, 8), block 32.
// ---------------------------------------------------------------------------
__global__ __launch_bounds__(32) void proj_gemm_kernel(
    const float* __restrict__ attn, const float* __restrict__ Wp,
    float* __restrict__ out) {
  int rt = blockIdx.x;
  int ct = blockIdx.y;
  int lane = threadIdx.x & 31;
  int lp = lane & 15, lh = lane >> 4;
  int row0 = rt * 16;
  int col0 = ct * 16;

  v8f c = {};
#pragma unroll
  for (int kc = 0; kc < 4; ++kc) {
    v16h a = load_a_f32(attn + (size_t)row0 * DMODEL, DMODEL, lp, lh, kc * 32, 1.0f);
    v16h b = load_b_f32(Wp, DMODEL, col0 + lp, lh, kc * 32);
    c = wmma_f16(a, b, c);
  }
#pragma unroll
  for (int i = 0; i < 8; ++i) {
    size_t r = (size_t)row0 + i + lh * 8;
    out[r * DMODEL + col0 + lp] = c[i];
  }
}

extern "C" void kernel_launch(void* const* d_in, const int* in_sizes, int n_in,
                              void* d_out, int out_size, void* d_ws, size_t ws_size,
                              hipStream_t stream) {
  (void)in_sizes; (void)n_in; (void)out_size; (void)ws_size;
  const float* x      = (const float*)d_in[0];
  const float* past_k = (const float*)d_in[1];
  const float* past_v = (const float*)d_in[2];
  const float* W_qkv  = (const float*)d_in[3];
  const float* W_proj = (const float*)d_in[4];

  const size_t OUT_ELEMS = (size_t)B_SZ * T_SZ * DMODEL;          // 4,194,304
  const size_t KV_ELEMS  = (size_t)B_SZ * NHEAD * TFULL * DHEAD;  // 8,388,608

  float* out   = (float*)d_out;
  float* k_out = out + OUT_ELEMS;
  float* v_out = k_out + KV_ELEMS;

  float* q_s  = (float*)d_ws;            // (B,H,T,Dh) fp32
  float* attn = q_s + OUT_ELEMS;         // (B,T,D) fp32

  // 1) splice past KV into output slabs (1,048,576 float4 per tensor)
  copy_past_kernel<<<dim3(4096, 2), 256, 0, stream>>>(
      (const float4*)past_k, (const float4*)past_v,
      (float4*)k_out, (float4*)v_out);

  // 2) qkv projection + scatter
  qkv_gemm_kernel<<<dim3(2048, 24), 32, 0, stream>>>(x, W_qkv, q_s, k_out, v_out);

  // 3) flash attention
  attention_kernel<<<dim3(512, 16), 32, 0, stream>>>(q_s, k_out, v_out, attn);

  // 4) output projection
  proj_gemm_kernel<<<dim3(2048, 8), 32, 0, stream>>>(attn, W_proj, out);
}
